// TopKSAESimple_56753697849561
// MI455X (gfx1250) — matrix operand order repaired and checked by reading, not compiled, not run on hardware
//
#include <hip/hip_runtime.h>
#include <stdint.h>

// Problem constants (fixed by the reference).
#define N_DIM 8192
#define D_DIM 1024
#define F_DIM 32768
#define K_TOP 64

// Fused encode kernel tiling: 32 rows (2 M-tiles/wave), 8 waves x 16 cols.
#define ROWS 32
#define WAVES 8
#define THREADS 256
#define COLS_PER_IT 128
#define ITERS (F_DIM / COLS_PER_IT)     // 256
#define KSTEPS (D_DIM / 32)             // 32 WMMA K-steps
#define XPAD 1032                       // padded LDS row stride (16B aligned, bank-friendly)
#define PENDCAP 1024
#define CPT 16                          // top-k candidates per thread per iteration

// Dynamic-LDS layout (bytes).
#define XA_OFF    0
#define XA_BYTES  (ROWS * XPAD * 2)                  // 66048: x-tile bf16
#define STG_OFF   (XA_OFF + XA_BYTES)
#define STG_BYTES (WAVES * ROWS * 16 * 4)            // 16384: latent staging
#define TKV_OFF   (STG_OFF + STG_BYTES)
#define TKV_BYTES (ROWS * K_TOP * 4)                 // 8192
#define TKI_OFF   (TKV_OFF + TKV_BYTES)
#define TKI_BYTES (ROWS * K_TOP * 4)                 // 8192
#define THR_OFF   (TKI_OFF + TKI_BYTES)
#define THR_BYTES (ROWS * 4)
#define PV_OFF    (THR_OFF + THR_BYTES)
#define PV_BYTES  (PENDCAP * 4)
#define PK_OFF    (PV_OFF + PV_BYTES)
#define PK_BYTES  (PENDCAP * 4)
#define CTL_OFF   (PK_OFF + PK_BYTES)
#define SMEM_TOTAL (CTL_OFF + 16)                    // ~107 KB < 320 KB/WGP

typedef __attribute__((ext_vector_type(16))) __bf16 v16bf;
typedef __attribute__((ext_vector_type(8)))  float  v8f;
typedef __attribute__((__vector_size__(4 * sizeof(int)))) int vint4;     // b128 payload type
typedef __attribute__((ext_vector_type(4))) float f32x4;

union FragBF { uint4 u2[2]; v16bf v; };

#define AS1 __attribute__((address_space(1)))
#define AS3 __attribute__((address_space(3)))

__device__ __forceinline__ unsigned short f2bf(float f) {
  union { float f; uint32_t u; } x; x.f = f;
  uint32_t r = x.u + 0x7FFFu + ((x.u >> 16) & 1u);   // round-to-nearest-even
  return (unsigned short)(r >> 16);
}

// ---------------- fp32 -> bf16 conversion ----------------
__global__ void k_cvt_bf16(const float* __restrict__ src, unsigned short* __restrict__ dst, long n) {
  long i = ((long)blockIdx.x * blockDim.x + threadIdx.x) * 8;
  long stride = (long)gridDim.x * blockDim.x * 8;
  for (; i < n; i += stride) {
    float4 a = *(const float4*)(src + i);
    float4 b = *(const float4*)(src + i + 4);
    union { unsigned short s[8]; uint4 v; } o;
    o.s[0] = f2bf(a.x); o.s[1] = f2bf(a.y); o.s[2] = f2bf(a.z); o.s[3] = f2bf(a.w);
    o.s[4] = f2bf(b.x); o.s[5] = f2bf(b.y); o.s[6] = f2bf(b.z); o.s[7] = f2bf(b.w);
    *(uint4*)(dst + i) = o.v;
  }
}

// ---------------- W_dec [D,F] -> W_decT [F,D] fp32 transpose ----------------
__global__ void k_transpose(const float* __restrict__ src, float* __restrict__ dst) {
  __shared__ float tile[32][33];
  const int fb = blockIdx.x * 32, db = blockIdx.y * 32;
  const int tx = threadIdx.x, ty = threadIdx.y;
  for (int j = ty; j < 32; j += 8)
    tile[j][tx] = src[(size_t)(db + j) * F_DIM + fb + tx];
  __syncthreads();
  for (int j = ty; j < 32; j += 8)
    dst[(size_t)(fb + j) * D_DIM + db + tx] = tile[tx][j];
}

// ---------------- zero-fill the 1 GB sparse output (non-temporal: keep L2 for W) ----------------
__global__ void k_zero(float* __restrict__ p, size_t n4) {
  size_t i = (size_t)blockIdx.x * blockDim.x + threadIdx.x;
  size_t stride = (size_t)gridDim.x * blockDim.x;
  f32x4 z = {0.f, 0.f, 0.f, 0.f};
  f32x4* pv = (f32x4*)p;
  for (; i < n4; i += stride) __builtin_nontemporal_store(z, pv + i);
}

// ---------------- fused WMMA encode + running top-64 + scatter ----------------
__launch_bounds__(THREADS)
__global__ void k_encode_topk(const unsigned short* __restrict__ Xb,   // [N,D] bf16
                              const unsigned short* __restrict__ Wb,   // [F,D] bf16
                              const float* __restrict__ b_enc,         // [F]
                              float* __restrict__ sparse,              // [N,F] (pre-zeroed)
                              float* __restrict__ tk_val,              // [N,K]
                              int*   __restrict__ tk_idx) {            // [N,K]
  extern __shared__ char smem[];
  unsigned short* xA  = (unsigned short*)(smem + XA_OFF);
  float*          stg = (float*)(smem + STG_OFF);
  float*          tkv = (float*)(smem + TKV_OFF);
  int*            tki = (int*)(smem + TKI_OFF);
  float*          thr = (float*)(smem + THR_OFF);
  float*          pvv = (float*)(smem + PV_OFF);
  int*            pkk = (int*)(smem + PK_OFF);
  int*            pend_cnt  = (int*)(smem + CTL_OFF);
  int*            left_flag = (int*)(smem + CTL_OFF + 4);

  const int tid  = threadIdx.x;
  const int wave = tid >> 5;
  const int lane = tid & 31;
  const int row0 = blockIdx.x * ROWS;

  // Load x-tile (32 x 1024 bf16 = 64 KB) into LDS via CDNA5 async copy when available.
  {
    const uint4* src = (const uint4*)(Xb + (size_t)row0 * D_DIM);   // 4096 uint4
#if __has_builtin(__builtin_amdgcn_global_load_async_to_lds_b128)
    for (int q = tid; q < (ROWS * D_DIM) / 8; q += THREADS) {
      int r = (q * 8) / D_DIM, c = (q * 8) % D_DIM;
      __builtin_amdgcn_global_load_async_to_lds_b128(
          (vint4*)(src + q), (AS3 vint4*)(xA + r * XPAD + c), 0, 0);
    }
#if __has_builtin(__builtin_amdgcn_s_wait_asynccnt)
    __builtin_amdgcn_s_wait_asynccnt(0);
#else
    asm volatile("s_wait_asynccnt 0x0" ::: "memory");
#endif
#else
    for (int q = tid; q < (ROWS * D_DIM) / 8; q += THREADS) {
      int r = (q * 8) / D_DIM, c = (q * 8) % D_DIM;
      *(uint4*)(xA + r * XPAD + c) = src[q];
    }
#endif
  }
  if (tid == 0) { *pend_cnt = 0; *left_flag = 0; }
  if (tid < ROWS) thr[tid] = -3.4e38f;
  __syncthreads();

  // Owner-private top-k state (thread tid<32 owns row tid).
  int   own_cnt = 0;
  float own_min = -3.4e38f;
  int   own_minpos = 0;

  const int lhalf = (lane < 16) ? 0 : 8;    // A-fragment K sub-offset per half-wave
  const int kbB   = (lane < 16) ? 0 : 8;    // B-fragment K sub-offset per half-wave
  const int nloc  = lane & 15;

  for (int it = 0; it < ITERS; ++it) {
    if (tid == 0) { *pend_cnt = 0; *left_flag = 0; }   // safe: everyone is past last drain

    const int fcol = it * COLS_PER_IT + wave * 16 + nloc;  // this lane's latent column
    const unsigned short* wrow = Wb + (size_t)fcol * D_DIM;
    if (fcol + COLS_PER_IT < F_DIM)
      __builtin_prefetch(Wb + (size_t)(fcol + COLS_PER_IT) * D_DIM, 0, 0);

    // --- GEMM: one B fragment feeds two M-tiles (rows 0-15 and 16-31). ---
    v8f acc0 = {0.f, 0.f, 0.f, 0.f, 0.f, 0.f, 0.f, 0.f};
    v8f acc1 = {0.f, 0.f, 0.f, 0.f, 0.f, 0.f, 0.f, 0.f};
    #pragma unroll 2
    for (int kk = 0; kk < KSTEPS; ++kk) {
      FragBF fa0, fa1, fb;
      const int ko = kk * 32 + lhalf;   // A: K = {ko..ko+7, ko+16..ko+23}
      fa0.u2[0] = *(const uint4*)(xA + nloc * XPAD + ko);
      fa0.u2[1] = *(const uint4*)(xA + nloc * XPAD + ko + 16);
      fa1.u2[0] = *(const uint4*)(xA + (16 + nloc) * XPAD + ko);
      fa1.u2[1] = *(const uint4*)(xA + (16 + nloc) * XPAD + ko + 16);
      const int kb = kk * 32 + kbB;     // B: K = {kb..kb+7, kb+16..kb+23}
      fb.u2[0] = *(const uint4*)(wrow + kb);
      fb.u2[1] = *(const uint4*)(wrow + kb + 16);
      acc0 = __builtin_amdgcn_wmma_f32_16x16x32_bf16(false, fa0.v, false, fb.v,
                                                     (short)0, acc0, false, false);
      acc1 = __builtin_amdgcn_wmma_f32_16x16x32_bf16(false, fa1.v, false, fb.v,
                                                     (short)0, acc1, false, false);
    }
    const float be = b_enc[fcol];
    #pragma unroll
    for (int r = 0; r < 8; ++r) { acc0[r] += be; acc1[r] += be; }

    // Stage 32x16 tile: C-layout VGPR r -> row (r + lhalf), col nloc.
    #pragma unroll
    for (int r = 0; r < 8; ++r) {
      stg[(wave * ROWS + (r + lhalf)) * 16 + nloc]        = acc0[r];
      stg[(wave * ROWS + (16 + r + lhalf)) * 16 + nloc]   = acc1[r];
    }
    __syncthreads();   // stage + pend reset visible

    // --- Top-k: overflow-robust push/drain rounds. ---
    const int m  = tid >> 3;            // row this thread filters (0..31)
    const int c0 = (tid & 7) * CPT;     // first of 16 columns
    int ci = 0;
    for (;;) {
      while (ci < CPT) {
        const int c = c0 + ci;
        float v = stg[((c >> 4) * ROWS + m) * 16 + (c & 15)];
        if (v > thr[m]) {
          int p = atomicAdd(pend_cnt, 1);
          if (p >= PENDCAP) break;                       // list full: retry this one next round
          pvv[p] = v; pkk[p] = (m << 16) | (it * COLS_PER_IT + c);
        }
        ++ci;
      }
      __syncthreads();                                   // pushes visible to owners
      if (tid < ROWS) {                                  // owners drain their row
        const int r = tid;
        int np = *pend_cnt; if (np > PENDCAP) np = PENDCAP;
        for (int p = 0; p < np; ++p) {
          if ((pkk[p] >> 16) == r) {
            float v = pvv[p];
            if (own_cnt < K_TOP) {
              tkv[r * K_TOP + own_cnt] = v; tki[r * K_TOP + own_cnt] = pkk[p] & 0xFFFF;
              if (++own_cnt == K_TOP) {
                own_min = tkv[r * K_TOP]; own_minpos = 0;
                for (int j = 1; j < K_TOP; ++j)
                  if (tkv[r * K_TOP + j] < own_min) { own_min = tkv[r * K_TOP + j]; own_minpos = j; }
              }
            } else if (v > own_min) {
              tkv[r * K_TOP + own_minpos] = v; tki[r * K_TOP + own_minpos] = pkk[p] & 0xFFFF;
              own_min = tkv[r * K_TOP]; own_minpos = 0;
              for (int j = 1; j < K_TOP; ++j)
                if (tkv[r * K_TOP + j] < own_min) { own_min = tkv[r * K_TOP + j]; own_minpos = j; }
            }
          }
        }
        thr[r] = (own_cnt < K_TOP) ? -3.4e38f : own_min;
      }
      if (ci < CPT) *left_flag = 1;                      // benign write race (same value)
      __syncthreads();                                   // drain + flag visible
      if (!*left_flag) break;
      if (tid == 0) { *pend_cnt = 0; *left_flag = 0; }
      __syncthreads();                                   // reset visible before next round
    }
  }

  // Scatter ReLU'd top-k into sparse output + save (val,idx) for decode.
  for (int e = tid; e < ROWS * K_TOP; e += THREADS) {
    const int r = e >> 6, j = e & 63;
    float v = tkv[r * K_TOP + j];
    int   c = tki[r * K_TOP + j];
    float rv = v > 0.f ? v : 0.f;
    sparse[(size_t)(row0 + r) * F_DIM + c] = rv;
    tk_val[(size_t)(row0 + r) * K_TOP + j] = rv;
    tk_idx[(size_t)(row0 + r) * K_TOP + j] = c;
  }
}

// ---------------- sparse decode: recon[n,:] = sum_j val_j * W_decT[idx_j,:] + b_dec ----------------
__global__ void k_decode(const float* __restrict__ tk_val, const int* __restrict__ tk_idx,
                         const float* __restrict__ Wt,      // [F,D] fp32 (transposed W_dec)
                         const float* __restrict__ b_dec,   // [D]
                         float* __restrict__ recon) {       // [N,D]
  const int n = blockIdx.x;
  const int tid = threadIdx.x;   // 256 threads, each owns 4 d-slots
  __shared__ float sv[K_TOP];
  __shared__ int   si[K_TOP];
  if (tid < K_TOP) {
    sv[tid] = tk_val[(size_t)n * K_TOP + tid];
    si[tid] = tk_idx[(size_t)n * K_TOP + tid];
  }
  __syncthreads();
  float acc[4];
  #pragma unroll
  for (int j = 0; j < 4; ++j) acc[j] = b_dec[tid + j * 256];
  for (int t = 0; t < K_TOP; ++t) {
    const float v = sv[t];
    const float* wr = Wt + (size_t)si[t] * D_DIM;
    #pragma unroll
    for (int j = 0; j < 4; ++j) acc[j] += v * wr[tid + j * 256];
  }
  #pragma unroll
  for (int j = 0; j < 4; ++j) recon[(size_t)n * D_DIM + tid + j * 256] = acc[j];
}

extern "C" void kernel_launch(void* const* d_in, const int* in_sizes, int n_in,
                              void* d_out, int out_size, void* d_ws, size_t ws_size,
                              hipStream_t stream) {
  (void)in_sizes; (void)n_in; (void)out_size; (void)ws_size;
  const float* x     = (const float*)d_in[0];
  const float* W_enc = (const float*)d_in[1];
  const float* b_enc = (const float*)d_in[2];
  const float* W_dec = (const float*)d_in[3];
  const float* b_dec = (const float*)d_in[4];
  // d_in[5] = k (compiled in as 64)

  // Workspace layout (~212 MB): x_bf16 | W_bf16 | W_decT | tk_val | tk_idx
  char* ws = (char*)d_ws;
  unsigned short* xb = (unsigned short*)ws;
  unsigned short* wb = (unsigned short*)(ws + (size_t)N_DIM * D_DIM * 2);
  float* wt  = (float*)(ws + (size_t)N_DIM * D_DIM * 2 + (size_t)F_DIM * D_DIM * 2);
  float* tkv = (float*)((char*)wt + (size_t)F_DIM * D_DIM * 4);
  int*   tki = (int*)((char*)tkv + (size_t)N_DIM * K_TOP * 4);

  float* recon  = (float*)d_out;                       // [N,D]
  float* sparse = recon + (size_t)N_DIM * D_DIM;       // [N,F]

  k_cvt_bf16<<<1024, 256, 0, stream>>>(x, xb, (long)N_DIM * D_DIM);
  k_cvt_bf16<<<4096, 256, 0, stream>>>(W_enc, wb, (long)F_DIM * D_DIM);
  k_transpose<<<dim3(F_DIM / 32, D_DIM / 32), dim3(32, 8), 0, stream>>>(W_dec, wt);
  k_zero<<<8192, 256, 0, stream>>>((float*)sparse, (size_t)N_DIM * F_DIM / 4);
  k_encode_topk<<<N_DIM / ROWS, THREADS, SMEM_TOTAL, stream>>>(xb, wb, b_enc, sparse, tkv, tki);
  k_decode<<<N_DIM, 256, 0, stream>>>(tkv, tki, wt, b_dec, recon);
}
